// PositionModulatedDeformableAttentionBlock_77163382440588
// MI455X (gfx1250) — matrix operand order repaired and checked by reading, not compiled, not run on hardware
//
#include <hip/hip_runtime.h>

#define EMB    256
#define HEADS  8
#define DHEAD  32
#define LEVELS 4
#define POINTS 4
#define BATCH  2
#define SEQ    21760
#define ROWS   (BATCH * SEQ)   // 43520
#define MTILES (ROWS / 16)     // 2720

typedef __attribute__((ext_vector_type(16))) _Float16 v16h;
typedef __attribute__((ext_vector_type(8)))  float    v8f;

// ---- WMMA fragment loaders (CDNA5 16-bit layouts, wave32) -------------------
// A (16x32 f16): lanes 0-15 -> M=lane, components i -> K = 16*g + i (contiguous, matches B)
__device__ __forceinline__ v16h wmma_load_a(const _Float16* ldsA, int lane, int kb, int ldk) {
  int m = lane & 15, g = lane >> 4;
  union { v16h h; unsigned u[8]; } f;
  const unsigned* base = (const unsigned*)(ldsA + m * ldk);
#pragma unroll
  for (int j = 0; j < 8; ++j) f.u[j] = base[(kb + 16 * g + 2 * j) >> 1];
  return f.h;
}

// B (32x16 f16), staged transposed in LDS as [N][K=32]: lane n = n0+(lane&15),
// components i -> K = 16*g + i  (contiguous: 8 dword loads)
__device__ __forceinline__ v16h wmma_load_b(const _Float16* ldsBT, int lane, int ldk, int n0) {
  int n = n0 + (lane & 15), g = lane >> 4;
  union { v16h h; unsigned u[8]; } f;
  const unsigned* base = (const unsigned*)(ldsBT + n * ldk + 16 * g);
#pragma unroll
  for (int j = 0; j < 8; ++j) f.u[j] = base[j];
  return f.h;
}

// ---- Kernel 1: value projection  v = mask0(img_feat @ W_value + b_value) ----
// grid (2720, 2), block 256: 16 rows x 128 cols per block, 16x16 tile per wave
__global__ __launch_bounds__(256) void vproj_kernel(
    const float* __restrict__ img, const unsigned char* __restrict__ mask,
    const float* __restrict__ Wv, const float* __restrict__ bv,
    float* __restrict__ vout) {
  __shared__ _Float16 Ah[16 * EMB];     // 8 KB
  __shared__ _Float16 Bc[128 * 32];     // 8 KB
  int t = threadIdx.x, lane = t & 31, wave = t >> 5;
  int row0 = blockIdx.x * 16;
  int nbase = blockIdx.y * 128;
  {
    int r = t >> 4, e0 = (t & 15) * 16;
    const float* a = img + (size_t)(row0 + r) * EMB + e0;
#pragma unroll
    for (int i = 0; i < 16; ++i) Ah[r * EMB + e0 + i] = (_Float16)a[i];
  }
  v8f c = {};
  for (int kb = 0; kb < EMB; kb += 32) {
    __syncthreads();
    {
      int nl = t >> 1, kk0 = (t & 1) * 16;
#pragma unroll
      for (int kk = 0; kk < 16; ++kk)
        Bc[nl * 32 + kk0 + kk] = (_Float16)Wv[(size_t)(kb + kk0 + kk) * EMB + nbase + nl];
    }
    __syncthreads();
    v16h a = wmma_load_a(Ah, lane, kb, EMB);
    v16h b = wmma_load_b(Bc, lane, 32, wave * 16);
    c = __builtin_amdgcn_wmma_f32_16x16x32_f16(false, a, false, b, (short)0, c, false, false);
  }
  int n = nbase + wave * 16 + (lane & 15), g = lane >> 4;
#pragma unroll
  for (int r = 0; r < 8; ++r) {
    int m = r + 8 * g, row = row0 + m;
    float val = c[r] + bv[n];
    if (mask[row]) val = 0.f;
    vout[(size_t)row * EMB + n] = val;
  }
}

// ---- Kernel 2: q = feat+pos; [off|attn] GEMM (N=384); softmax; bilinear sample
__global__ __launch_bounds__(256) void qside_kernel(
    const float* __restrict__ qf, const float* __restrict__ qp,
    const float* __restrict__ qpts, const float* __restrict__ vr,
    const float* __restrict__ Woff, const float* __restrict__ boff,
    const float* __restrict__ Wattn, const float* __restrict__ battn,
    const float* __restrict__ vbuf, float* __restrict__ outh) {
  __shared__ _Float16 Ah[16 * EMB];     // 8 KB
  __shared__ _Float16 Bc[384 * 32];     // 24 KB
  __shared__ float    Res[16 * 384];    // 24 KB: [0,256)=raw offsets, [256,384)=logits->probs
  int t = threadIdx.x, lane = t & 31, wave = t >> 5;
  int row0 = blockIdx.x * 16;
  {
    int r = t >> 4, e0 = (t & 15) * 16;
    const float* a = qf + (size_t)(row0 + r) * EMB + e0;
    const float* b = qp + (size_t)(row0 + r) * EMB + e0;
#pragma unroll
    for (int i = 0; i < 16; ++i) Ah[r * EMB + e0 + i] = (_Float16)(a[i] + b[i]);
  }
  v8f c0 = {}, c1 = {}, c2 = {};
  for (int kb = 0; kb < EMB; kb += 32) {
    __syncthreads();
    for (int nl = t; nl < 384; nl += 256) {
      if (nl < 256) {
#pragma unroll
        for (int kk = 0; kk < 32; ++kk)
          Bc[nl * 32 + kk] = (_Float16)Woff[(size_t)(kb + kk) * 256 + nl];
      } else {
        int n2 = nl - 256;
#pragma unroll
        for (int kk = 0; kk < 32; ++kk)
          Bc[nl * 32 + kk] = (_Float16)Wattn[(size_t)(kb + kk) * 128 + n2];
      }
    }
    __syncthreads();
    v16h a = wmma_load_a(Ah, lane, kb, EMB);
    v16h b0 = wmma_load_b(Bc, lane, 32, (wave * 3 + 0) * 16);
    c0 = __builtin_amdgcn_wmma_f32_16x16x32_f16(false, a, false, b0, (short)0, c0, false, false);
    v16h b1 = wmma_load_b(Bc, lane, 32, (wave * 3 + 1) * 16);
    c1 = __builtin_amdgcn_wmma_f32_16x16x32_f16(false, a, false, b1, (short)0, c1, false, false);
    v16h b2 = wmma_load_b(Bc, lane, 32, (wave * 3 + 2) * 16);
    c2 = __builtin_amdgcn_wmma_f32_16x16x32_f16(false, a, false, b2, (short)0, c2, false, false);
  }
  {
    int nc = lane & 15, g = lane >> 4;
#pragma unroll
    for (int r = 0; r < 8; ++r) {
      int m = r + 8 * g;
      Res[m * 384 + (wave * 3 + 0) * 16 + nc] = c0[r];
      Res[m * 384 + (wave * 3 + 1) * 16 + nc] = c1[r];
      Res[m * 384 + (wave * 3 + 2) * 16 + nc] = c2[r];
    }
  }
  __syncthreads();
  if (t < 128) {   // softmax over L*P=16, in place: logits -> probs
    int ql = t >> 3, h = t & 7;
    float lg[16], mx = -1e30f;
#pragma unroll
    for (int i = 0; i < 16; ++i) {
      lg[i] = Res[ql * 384 + 256 + h * 16 + i] + battn[h * 16 + i];
      mx = fmaxf(mx, lg[i]);
    }
    float s = 0.f;
#pragma unroll
    for (int i = 0; i < 16; ++i) { lg[i] = __expf(lg[i] - mx); s += lg[i]; }
    float inv = 1.f / s;
#pragma unroll
    for (int i = 0; i < 16; ++i) Res[ql * 384 + 256 + h * 16 + i] = lg[i] * inv;
  }
  __syncthreads();
  // sampling: thread -> (q_local = t>>4, h = (t>>1)&7, half-head dh = t&1)
  int ql = t >> 4, h = (t >> 1) & 7, dh = t & 1;
  int row = row0 + ql;
  int bi = row / SEQ;
  float refx = qpts[(size_t)row * 2 + 0], refy = qpts[(size_t)row * 2 + 1];
  float acc[16];
#pragma unroll
  for (int i = 0; i < 16; ++i) acc[i] = 0.f;
  const int LH[4] = {128, 64, 32, 16};
  const int LW[4] = {128, 64, 32, 16};
  const int LST[4] = {0, 16384, 20480, 21504};
#pragma unroll
  for (int l = 0; l < LEVELS; ++l) {
    float rx = refx * vr[(bi * LEVELS + l) * 2 + 0];
    float ry = refy * vr[(bi * LEVELS + l) * 2 + 1];
    int W = LW[l], H = LH[l], st = LST[l];
#pragma unroll
    for (int p = 0; p < POINTS; ++p) {
      int oi = ((h * LEVELS + l) * POINTS + p) * 2;
      float ox = Res[ql * 384 + oi] + boff[oi];
      float oy = Res[ql * 384 + oi + 1] + boff[oi + 1];
      float x = rx * (float)W + ox - 0.5f;
      float y = ry * (float)H + oy - 0.5f;
      float x0f = floorf(x), y0f = floorf(y);
      float wx = x - x0f, wy = y - y0f;
      int x0 = (int)x0f, y0 = (int)y0f;
      float aw = Res[ql * 384 + 256 + h * 16 + l * POINTS + p];
#pragma unroll
      for (int dy = 0; dy < 2; ++dy) {
#pragma unroll
        for (int dx = 0; dx < 2; ++dx) {
          int xi = x0 + dx, yi = y0 + dy;
          float w = (dx ? wx : 1.f - wx) * (dy ? wy : 1.f - wy) * aw;
          if (xi < 0 || xi >= W || yi < 0 || yi >= H) w = 0.f;
          int xc = xi < 0 ? 0 : (xi > W - 1 ? W - 1 : xi);
          int yc = yi < 0 ? 0 : (yi > H - 1 ? H - 1 : yi);
          const float4* vp = (const float4*)(vbuf +
              ((size_t)(bi * SEQ + st + yc * W + xc)) * EMB + h * DHEAD + dh * 16);
#pragma unroll
          for (int j = 0; j < 4; ++j) {
            float4 q4 = vp[j];
            acc[4 * j + 0] += w * q4.x; acc[4 * j + 1] += w * q4.y;
            acc[4 * j + 2] += w * q4.z; acc[4 * j + 3] += w * q4.w;
          }
        }
      }
    }
  }
  float* o = outh + (size_t)row * EMB + h * DHEAD + dh * 16;
#pragma unroll
  for (int i = 0; i < 16; ++i) o[i] = acc[i];
}

// ---- Kernel 3: update = out @ W_out + b_out; x = qf + update; LayerNorm(x) ---
// Reads its own 16 rows of io (=d_out) before overwriting them.
__global__ __launch_bounds__(256) void oproj_ln_kernel(
    const float* __restrict__ qf, const float* __restrict__ Wo,
    const float* __restrict__ bo, const float* __restrict__ gamma,
    const float* __restrict__ beta, float* __restrict__ io) {
  __shared__ _Float16 Ah[16 * EMB];     // 8 KB
  __shared__ _Float16 Bc[256 * 32];     // 16 KB
  __shared__ float    Res[16 * EMB];    // 16 KB
  int t = threadIdx.x, lane = t & 31, wave = t >> 5;
  int row0 = blockIdx.x * 16;
  {
    int r = t >> 4, e0 = (t & 15) * 16;
    const float* a = io + (size_t)(row0 + r) * EMB + e0;
#pragma unroll
    for (int i = 0; i < 16; ++i) Ah[r * EMB + e0 + i] = (_Float16)a[i];
  }
  v8f c0 = {}, c1 = {};
  for (int kb = 0; kb < EMB; kb += 32) {
    __syncthreads();
    {
      int nl = t;
#pragma unroll
      for (int kk = 0; kk < 32; ++kk)
        Bc[nl * 32 + kk] = (_Float16)Wo[(size_t)(kb + kk) * EMB + nl];
    }
    __syncthreads();
    v16h a = wmma_load_a(Ah, lane, kb, EMB);
    v16h b0 = wmma_load_b(Bc, lane, 32, (wave * 2 + 0) * 16);
    c0 = __builtin_amdgcn_wmma_f32_16x16x32_f16(false, a, false, b0, (short)0, c0, false, false);
    v16h b1 = wmma_load_b(Bc, lane, 32, (wave * 2 + 1) * 16);
    c1 = __builtin_amdgcn_wmma_f32_16x16x32_f16(false, a, false, b1, (short)0, c1, false, false);
  }
  {
    int nc = lane & 15, g = lane >> 4;
#pragma unroll
    for (int r = 0; r < 8; ++r) {
      int m = r + 8 * g, row = row0 + m;
      int n0 = (wave * 2 + 0) * 16 + nc, n1 = (wave * 2 + 1) * 16 + nc;
      Res[m * EMB + n0] = c0[r] + bo[n0] + qf[(size_t)row * EMB + n0];
      Res[m * EMB + n1] = c1[r] + bo[n1] + qf[(size_t)row * EMB + n1];
    }
  }
  __syncthreads();
#pragma unroll
  for (int rr = 0; rr < 2; ++rr) {
    int m = wave * 2 + rr, row = row0 + m;
    float xv[8], s = 0.f, s2 = 0.f;
#pragma unroll
    for (int j = 0; j < 8; ++j) {
      xv[j] = Res[m * EMB + lane * 8 + j];
      s += xv[j]; s2 += xv[j] * xv[j];
    }
#pragma unroll
    for (int off = 16; off > 0; off >>= 1) {
      s  += __shfl_xor(s, off, 32);
      s2 += __shfl_xor(s2, off, 32);
    }
    float mu = s * (1.f / EMB);
    float var = s2 * (1.f / EMB) - mu * mu;
    float inv = rsqrtf(var + 1e-5f);
#pragma unroll
    for (int j = 0; j < 8; ++j) {
      int e = lane * 8 + j;
      io[(size_t)row * EMB + e] = (xv[j] - mu) * inv * gamma[e] + beta[e];
    }
  }
}

extern "C" void kernel_launch(void* const* d_in, const int* in_sizes, int n_in,
                              void* d_out, int out_size, void* d_ws, size_t ws_size,
                              hipStream_t stream) {
  const float*         img_feat = (const float*)d_in[0];
  /* d_in[1] img_shapes (compile-time constants) */
  const unsigned char* img_mask = (const unsigned char*)d_in[2];
  const float*         vr       = (const float*)d_in[3];
  const float*         qf       = (const float*)d_in[4];
  const float*         qp       = (const float*)d_in[5];
  const float*         qpts     = (const float*)d_in[6];
  const float*         Wv       = (const float*)d_in[7];
  const float*         bv       = (const float*)d_in[8];
  const float*         Woff     = (const float*)d_in[9];
  const float*         boff     = (const float*)d_in[10];
  const float*         Wattn    = (const float*)d_in[11];
  const float*         battn    = (const float*)d_in[12];
  const float*         Wo       = (const float*)d_in[13];
  const float*         bo       = (const float*)d_in[14];
  const float*         gamma    = (const float*)d_in[15];
  const float*         beta     = (const float*)d_in[16];
  float* out  = (float*)d_out;
  float* vbuf = (float*)d_ws;   // 43520*256 floats = 44.6 MB (fits L2)

  vproj_kernel<<<dim3(MTILES, 2), 256, 0, stream>>>(img_feat, img_mask, Wv, bv, vbuf);
  qside_kernel<<<dim3(MTILES), 256, 0, stream>>>(qf, qp, qpts, vr, Woff, boff,
                                                 Wattn, battn, vbuf, out);
  oproj_ln_kernel<<<dim3(MTILES), 256, 0, stream>>>(qf, Wo, bo, gamma, beta, out);
}
